// CirculantSTRING_13700945674724
// MI455X (gfx1250) — matrix operand order repaired and checked by reading, not compiled, not run on hardware
//
#include <hip/hip_runtime.h>
#include <hip/hip_bf16.h>

// ---------------------------------------------------------------------------
// Circulant phase filter:  out[b,n,:] = irfft( exp(i*u[n,:]) * rfft(x[b,n,:]) )
//   u[n,f] = pos[n,0]*v0[f] + pos[n,1]*v1[f],  v_k = 2*Im(rfft(circ[k]))
// Fused DFT-GEMMs on the CDNA5 f16 WMMA pipe (v_wmma_f32_16x16x32_f16) with
// an f16x3 error-compensated split (hi*hi + hi*lo + lo*hi, f32 accumulate)
// => ~fp32 accuracy at f16 matrix-op rates.
//   B=32, N=1024, D=768, F=385 (padded to FP=448 = 7 chunks of 64)
// ---------------------------------------------------------------------------

#define D_DIM   768
#define FP      448           // padded frequency bins (zeroed beyond 385)
#define FC      64            // frequency chunk (4 tiles of 16)
#define NCH     (FP / FC)     // 7 chunks
#define ROWS    32            // rows (b,n) per workgroup
#define XS_STR  776           // LDS stride (halves) for X planes
#define ZS_STR  72            // LDS stride (halves) for Z planes
#define NFT_W   (FP / 16)     // 28 f-tiles for forward W
#define NKT_W   (D_DIM / 32)  // 24 k-tiles for forward W
#define NKT_B   (FP / 32)     // 14 k-tiles for inverse B
#define NDT_B   (D_DIM / 16)  // 48 d-tiles for inverse B
#define TW      (NKT_W * NFT_W * 512)   // 344064 halves per W table
#define TB      (NKT_B * NDT_B * 512)   // 344064 halves per B table
#define TWO_PI_OVER_D 0.008181230868723419f  // 2*pi/768

typedef _Float16 half_t;
typedef __attribute__((ext_vector_type(4)))  _Float16 v4h;
typedef __attribute__((ext_vector_type(8)))  _Float16 v8h;
typedef __attribute__((ext_vector_type(16))) _Float16 v16h;
typedef __attribute__((ext_vector_type(8)))  float    v8f;

#define WMMA_F16(a, b, c) \
    __builtin_amdgcn_wmma_f32_16x16x32_f16(false, (a), false, (b), (short)0, (c), false, false)

// ---------------------------------------------------------------------------
// Table builders (angles reduced mod 768 before sin/cos => exact arguments)
// ---------------------------------------------------------------------------

// v_k[f] = 2*Im(rfft(circ[k]))[f] = -2 * sum_d circ[k][d] * sin(2*pi*f*d/D)
__global__ void circ_build_v(const float* __restrict__ circ,
                             float* __restrict__ v0, float* __restrict__ v1) {
    int f = blockIdx.x * 64 + threadIdx.x;
    if (f >= FP) return;
    float s0 = 0.0f, s1 = 0.0f;
    if (f < 385) {
        for (int d = 0; d < D_DIM; ++d) {
            int r = (int)(((long long)d * f) % D_DIM);
            float sn = sinf((float)r * TWO_PI_OVER_D);
            s0 += circ[d] * sn;            // circ[0][d]
            s1 += circ[D_DIM + d] * sn;    // circ[1][d]
        }
    }
    v0[f] = -2.0f * s0;
    v1[f] = -2.0f * s1;
}

// Per-row phase tables cos(u[n,f]), sin(u[n,f])  (kept in f32 for accuracy).
__global__ void circ_build_phase(const int* __restrict__ pos,
                                 const float* __restrict__ v0,
                                 const float* __restrict__ v1,
                                 float* __restrict__ cu, float* __restrict__ su) {
    int i = blockIdx.x * 256 + threadIdx.x;
    if (i >= 1024 * FP) return;
    int n = i / FP, f = i % FP;
    float u = (float)pos[n * 2 + 0] * v0[f] + (float)pos[n * 2 + 1] * v1[f];
    float s, c;
    sincosf(u, &s, &c);
    cu[i] = c;
    su[i] = s;
}

// Forward DFT W, pre-swizzled into the 32x16 f16 B-fragment lane layout:
//   swz[tile][lane][e]:  K = (lane>>4)*16 + e (within k-tile), N = lane&15.
// hi/lo planes give the f16x3 split of each f32 coefficient.
__global__ void circ_swz_w(half_t* __restrict__ Wrh, half_t* __restrict__ Wrl,
                           half_t* __restrict__ Wih, half_t* __restrict__ Wil) {
    int i = blockIdx.x * 256 + threadIdx.x;
    if (i >= TW) return;
    int tile = i >> 9, r = i & 511;
    int lane = r >> 4, e = r & 15;
    int kt = tile / NFT_W, ft = tile % NFT_W;
    int d = kt * 32 + (lane >> 4) * 16 + e;   // 0..767
    int f = ft * 16 + (lane & 15);            // 0..447
    int m = (int)(((long long)d * f) % D_DIM);
    float th = (float)m * TWO_PI_OVER_D;
    float s, c;
    sincosf(th, &s, &c);
    bool act = (f < 385);
    float wr = act ? c : 0.0f;
    float wi = act ? -s : 0.0f;
    half_t h;
    h = (half_t)wr; Wrh[i] = h; Wrl[i] = (half_t)(wr - (float)h);
    h = (half_t)wi; Wih[i] = h; Wil[i] = (half_t)(wi - (float)h);
}

// Inverse (irfft) matrix B, same pre-swizzled fragment layout.
__global__ void circ_swz_b(half_t* __restrict__ Brh, half_t* __restrict__ Brl,
                           half_t* __restrict__ Bih, half_t* __restrict__ Bil) {
    int i = blockIdx.x * 256 + threadIdx.x;
    if (i >= TB) return;
    int tile = i >> 9, r = i & 511;
    int lane = r >> 4, e = r & 15;
    int kt = tile / NDT_B, dt = tile % NDT_B;
    int f = kt * 32 + (lane >> 4) * 16 + e;   // 0..447
    int d = dt * 16 + (lane & 15);            // 0..767
    int m = (int)(((long long)d * f) % D_DIM);
    float th = (float)m * TWO_PI_OVER_D;
    float s, c;
    sincosf(th, &s, &c);
    bool act = (f < 385);
    // irfft weights: DC/Nyquist count once (imag ignored), others twice.
    float cf = (f == 0 || f == 384) ? 1.0f : (act ? 2.0f : 0.0f);
    float sf = (f == 0 || f == 384 || !act) ? 0.0f : 2.0f;
    float br = cf * c * (1.0f / (float)D_DIM);
    float bi = -sf * s * (1.0f / (float)D_DIM);
    half_t h;
    h = (half_t)br; Brh[i] = h; Brl[i] = (half_t)(br - (float)h);
    h = (half_t)bi; Bih[i] = h; Bil[i] = (half_t)(bi - (float)h);
}

// ---------------------------------------------------------------------------
// A-fragment loader: this kernel's contiguous-k 16x32 f16 A layout.
//   lanes 0-15 hold K = [0..15] of row M=lane&15,
//   lanes 16-31 hold K = [16..31].
// ---------------------------------------------------------------------------
__device__ __forceinline__ v16h load_a_frag(const half_t* rowp, int k0, int hk) {
    v8h lo = *(const v8h*)(rowp + k0 + hk * 16);
    v8h hi = *(const v8h*)(rowp + k0 + 8 + hk * 16);
    return __builtin_shufflevector(lo, hi, 0, 1, 2, 3, 4, 5, 6, 7,
                                   8, 9, 10, 11, 12, 13, 14, 15);
}

// ---------------------------------------------------------------------------
// Fused main kernel: per 32-row block,
//   Y = X*W (f16x3) -> phase rotate in registers -> split -> out += Z*B (f16x3)
// 8 waves / 256 threads; all GEMMs on v_wmma_f32_16x16x32_f16.
// ---------------------------------------------------------------------------
__global__ __launch_bounds__(256, 1)
void circ_main(const float* __restrict__ x,
               const half_t* __restrict__ Wrh, const half_t* __restrict__ Wrl,
               const half_t* __restrict__ Wih, const half_t* __restrict__ Wil,
               const half_t* __restrict__ Brh, const half_t* __restrict__ Brl,
               const half_t* __restrict__ Bih, const half_t* __restrict__ Bil,
               const float* __restrict__ cu, const float* __restrict__ su,
               float* __restrict__ out) {
    __shared__ half_t Xh[ROWS * XS_STR];   // X hi plane  ~49.7 KB
    __shared__ half_t Xl[ROWS * XS_STR];   // X lo plane  ~49.7 KB
    __shared__ half_t Zrh[ROWS * ZS_STR];  // Z planes (chunk)  4 x ~4.6 KB
    __shared__ half_t Zrl[ROWS * ZS_STR];
    __shared__ half_t Zih[ROWS * ZS_STR];
    __shared__ half_t Zil[ROWS * ZS_STR];

    const int tid  = threadIdx.x;
    const int wave = tid >> 5;     // 0..7
    const int lane = tid & 31;
    const int l15  = lane & 15;
    const int hk   = lane >> 4;

    const long long m0 = (long long)blockIdx.x * ROWS;   // first flat row
    const int n0 = (int)(m0 & 1023);                     // n index of row 0

    // ---- stage X into LDS as f16 hi/lo planes (float4 global reads) ----
    {
        const float4* xg = (const float4*)(x + m0 * D_DIM);
        for (int i = tid; i < ROWS * (D_DIM / 4); i += 256) {
            int r = i / (D_DIM / 4), c4 = i % (D_DIM / 4);
            float4 v = xg[r * (D_DIM / 4) + c4];
            v4h h, l;
            h[0] = (half_t)v.x; l[0] = (half_t)(v.x - (float)h[0]);
            h[1] = (half_t)v.y; l[1] = (half_t)(v.y - (float)h[1]);
            h[2] = (half_t)v.z; l[2] = (half_t)(v.z - (float)h[2]);
            h[3] = (half_t)v.w; l[3] = (half_t)(v.w - (float)h[3]);
            *(v4h*)&Xh[r * XS_STR + c4 * 4] = h;
            *(v4h*)&Xl[r * XS_STR + c4 * 4] = l;
        }
    }

    // ---- persistent output accumulators: 12 tiles per wave (32x768 total) ----
    v8f acc[12];
    v8f zero8 = {};
#pragma unroll
    for (int t = 0; t < 12; ++t) acc[t] = zero8;

    // Stage-A tile owned by this wave: (mt, ft) covers 32 x FC per chunk.
    const int a_mt  = wave & 1;     // 0..1 : M tile
    const int a_ft  = wave >> 1;    // 0..3 : F tile within chunk
    const half_t* arow_h = Xh + (a_mt * 16 + l15) * XS_STR;
    const half_t* arow_l = Xl + (a_mt * 16 + l15) * XS_STR;

    for (int ch = 0; ch < NCH; ++ch) {
        const int f0c = ch * FC;
        __syncthreads();  // Xs ready (ch==0) / prev chunk's Z fully consumed

        // ---- Stage A: Y(chunk) = X * W  via f16x3 split ----
        const int ftg = ch * 4 + a_ft;           // global f-tile 0..27
        v8f yr = zero8, yi = zero8;
#pragma unroll 4
        for (int kt = 0; kt < NKT_W; ++kt) {
            v16h ah = load_a_frag(arow_h, kt * 32, hk);
            v16h al = load_a_frag(arow_l, kt * 32, hk);
            const long tb = ((long)(kt * NFT_W + ftg)) * 512 + lane * 16;
            v16h wrh = *(const v16h*)(Wrh + tb);
            v16h wrl = *(const v16h*)(Wrl + tb);
            v16h wih = *(const v16h*)(Wih + tb);
            v16h wil = *(const v16h*)(Wil + tb);
            yr = WMMA_F16(ah, wrh, yr);
            yr = WMMA_F16(ah, wrl, yr);
            yr = WMMA_F16(al, wrh, yr);
            yi = WMMA_F16(ah, wih, yi);
            yi = WMMA_F16(ah, wil, yi);
            yi = WMMA_F16(al, wih, yi);
        }

        // prefetch next chunk's W tiles (gfx1250 global_prefetch path)
        if (ch + 1 < NCH && lane < NKT_W) {
            const long pb = ((long)(lane * NFT_W + (ch + 1) * 4 + a_ft)) * 512;
            __builtin_prefetch(Wrh + pb, 0, 1);
            __builtin_prefetch(Wrl + pb, 0, 1);
            __builtin_prefetch(Wih + pb, 0, 1);
            __builtin_prefetch(Wil + pb, 0, 1);
        }

        // ---- Stage B: phase rotate in registers, split to f16 hi/lo in LDS ----
#pragma unroll
        for (int j = 0; j < 8; ++j) {
            const int rrow = a_mt * 16 + j + 8 * hk;    // 0..31
            const int ccol = a_ft * 16 + l15;           // chunk-local 0..63
            const int pidx = (n0 + rrow) * FP + f0c + ccol;
            const float pc = cu[pidx], ps = su[pidx];
            const float zr = pc * yr[j] - ps * yi[j];
            const float zi = pc * yi[j] + ps * yr[j];
            const half_t zrh_ = (half_t)zr;
            const half_t zih_ = (half_t)zi;
            const int o = rrow * ZS_STR + ccol;
            Zrh[o] = zrh_; Zrl[o] = (half_t)(zr - (float)zrh_);
            Zih[o] = zih_; Zil[o] = (half_t)(zi - (float)zih_);
        }
        __syncthreads();

        // ---- Stage C: out += Zr*Br + Zi*Bi over this chunk (f16x3 each) ----
#pragma unroll
        for (int t = 0; t < 12; ++t) {
            const int g   = wave + t * 8;     // 0..95 output tiles
            const int mt2 = g & 1;
            const int dt  = g >> 1;           // 0..47
            const half_t* zrow_base = (const half_t*)0;
            const int zr_off = (mt2 * 16 + l15) * ZS_STR;
            v8f a8 = acc[t];
#pragma unroll
            for (int ks = 0; ks < 2; ++ks) {
                const int k0 = ks * 32;
                v16h fzrh = load_a_frag(Zrh + zr_off, k0, hk);
                v16h fzrl = load_a_frag(Zrl + zr_off, k0, hk);
                v16h fzih = load_a_frag(Zih + zr_off, k0, hk);
                v16h fzil = load_a_frag(Zil + zr_off, k0, hk);
                const long tb = ((long)((ch * 2 + ks) * NDT_B + dt)) * 512 + lane * 16;
                v16h fbrh = *(const v16h*)(Brh + tb);
                v16h fbrl = *(const v16h*)(Brl + tb);
                v16h fbih = *(const v16h*)(Bih + tb);
                v16h fbil = *(const v16h*)(Bil + tb);
                a8 = WMMA_F16(fzrh, fbrh, a8);
                a8 = WMMA_F16(fzrh, fbrl, a8);
                a8 = WMMA_F16(fzrl, fbrh, a8);
                a8 = WMMA_F16(fzih, fbih, a8);
                a8 = WMMA_F16(fzih, fbil, a8);
                a8 = WMMA_F16(fzil, fbih, a8);
            }
            acc[t] = a8;
            (void)zrow_base;
        }
    }

    // ---- store 32x768 block (each half-wave writes 16 consecutive floats) ----
    float* og = out + m0 * D_DIM;
#pragma unroll
    for (int t = 0; t < 12; ++t) {
        const int g   = wave + t * 8;
        const int mt2 = g & 1;
        const int d0  = (g >> 1) * 16;
#pragma unroll
        for (int j = 0; j < 8; ++j) {
            og[(mt2 * 16 + j + 8 * hk) * D_DIM + d0 + l15] = acc[t][j];
        }
    }
}

// ---------------------------------------------------------------------------
// Launcher. Workspace layout:
//   f32: cu,su (1024*448 each), v0,v1 (448 each)      = 3.67 MB
//   f16: Wrh,Wrl,Wih,Wil (344064 each)                = 2.75 MB
//        Brh,Brl,Bih,Bil (344064 each)                = 2.75 MB
//   total ~9.2 MB of d_ws
// ---------------------------------------------------------------------------
extern "C" void kernel_launch(void* const* d_in, const int* in_sizes, int n_in,
                              void* d_out, int out_size, void* d_ws, size_t ws_size,
                              hipStream_t stream) {
    const float* x    = (const float*)d_in[0];   // (32, 1024, 768) f32
    const float* circ = (const float*)d_in[1];   // (2, 768) f32
    const int*   pos  = (const int*)d_in[2];     // (1024, 2) i32
    float* out = (float*)d_out;                  // (32, 1024, 768) f32

    float* ws = (float*)d_ws;
    float* cu = ws;
    float* su = cu + 1024 * FP;
    float* v0 = su + 1024 * FP;
    float* v1 = v0 + FP;
    half_t* hb  = (half_t*)(v1 + FP);
    half_t* Wrh = hb;
    half_t* Wrl = Wrh + TW;
    half_t* Wih = Wrl + TW;
    half_t* Wil = Wih + TW;
    half_t* Brh = Wil + TW;
    half_t* Brl = Brh + TB;
    half_t* Bih = Brl + TB;
    half_t* Bil = Bih + TB;

    circ_build_v<<<(FP + 63) / 64, 64, 0, stream>>>(circ, v0, v1);
    circ_build_phase<<<(1024 * FP + 255) / 256, 256, 0, stream>>>(pos, v0, v1, cu, su);
    circ_swz_w<<<(TW + 255) / 256, 256, 0, stream>>>(Wrh, Wrl, Wih, Wil);
    circ_swz_b<<<(TB + 255) / 256, 256, 0, stream>>>(Brh, Brl, Bih, Bil);
    circ_main<<<1024, 256, 0, stream>>>(x, Wrh, Wrl, Wih, Wil,
                                        Brh, Brl, Bih, Bil, cu, su, out);
}
